// AntiSymmetric_Net_4320737100478
// MI455X (gfx1250) — matrix operand order repaired and checked by reading, not compiled, not run on hardware
//
#include <hip/hip_runtime.h>
#include <hip/hip_bf16.h>
#include <math.h>

typedef __attribute__((ext_vector_type(16))) _Float16 v16h;
typedef __attribute__((ext_vector_type(8)))  float    v8f;

#define AS_EPS   0.1f
#define AS_GAMMA 0.1f

// ---------------- WMMA fragment loads (f32 memory -> f16 fragments, b128 loads) ----
// A: activations, row-major [M,K]. Lane holds row lane%16.
//   slots 0..15 = K: k0 + 16*(lane/16) + 0..15 (contiguous)
__device__ inline v16h load_a_frag(const float* __restrict__ p) {
    float4 x0 = *(const float4*)(p);
    float4 x1 = *(const float4*)(p + 4);
    float4 x2 = *(const float4*)(p + 8);
    float4 x3 = *(const float4*)(p + 12);
    v16h f;
    f[0]  = (_Float16)x0.x; f[1]  = (_Float16)x0.y; f[2]  = (_Float16)x0.z; f[3]  = (_Float16)x0.w;
    f[4]  = (_Float16)x1.x; f[5]  = (_Float16)x1.y; f[6]  = (_Float16)x1.z; f[7]  = (_Float16)x1.w;
    f[8]  = (_Float16)x2.x; f[9]  = (_Float16)x2.y; f[10] = (_Float16)x2.z; f[11] = (_Float16)x2.w;
    f[12] = (_Float16)x3.x; f[13] = (_Float16)x3.y; f[14] = (_Float16)x3.z; f[15] = (_Float16)x3.w;
    return f;
}

// B: weights W row-major [Nd,K]; B[k,n] = W[n,k]. Lane holds column lane%16.
//   slots 0..15 = K: k0 + 16*(lane/16) + 0..15 (contiguous)
__device__ inline v16h load_b_frag(const float* __restrict__ p) {
    float4 x0 = *(const float4*)(p);
    float4 x1 = *(const float4*)(p + 4);
    float4 x2 = *(const float4*)(p + 8);
    float4 x3 = *(const float4*)(p + 12);
    v16h f;
    f[0]  = (_Float16)x0.x; f[1]  = (_Float16)x0.y; f[2]  = (_Float16)x0.z; f[3]  = (_Float16)x0.w;
    f[4]  = (_Float16)x1.x; f[5]  = (_Float16)x1.y; f[6]  = (_Float16)x1.z; f[7]  = (_Float16)x1.w;
    f[8]  = (_Float16)x2.x; f[9]  = (_Float16)x2.y; f[10] = (_Float16)x2.z; f[11] = (_Float16)x2.w;
    f[12] = (_Float16)x3.x; f[13] = (_Float16)x3.y; f[14] = (_Float16)x3.z; f[15] = (_Float16)x3.w;
    return f;
}

__device__ inline v8f wmma_f16(v16h a, v16h b, v8f c) {
    return __builtin_amdgcn_wmma_f32_16x16x32_f16(false, a, false, b, (short)0, c, false, false);
}

// ---------------- WMMA GEMM: out[M,Nd] = epilogue(A[M,K] @ W[Nd,K]^T) ----------------
// Each wave computes a 32x32 output block (2x2 WMMA register tiles, 4 wmma / K-step).
// EPI 0: relu(acc + bias[n])
// EPI 1: acc
// EPI 2: res + EPS * tanh(acc + add + bias[n])   (AntiSymmetric Euler step)
// EPI 3: acc + bias[n]
template <int EPI>
__device__ inline void store_tile(v8f acc, const float* __restrict__ bias,
                                  const float* __restrict__ add, const float* __restrict__ res,
                                  float* __restrict__ out, int Nd, int row0, int col) {
    const float bv = (EPI == 1) ? 0.f : bias[col];
    float* po = out + (size_t)row0 * Nd + col;
#pragma unroll
    for (int r = 0; r < 8; ++r) {
        const size_t idx = (size_t)r * Nd;
        float v = acc[r];
        if (EPI == 0)      { v += bv; v = v > 0.f ? v : 0.f; }
        else if (EPI == 2) { v = res[(size_t)(row0 + r) * Nd + col]
                                 + AS_EPS * tanhf(v + add[(size_t)(row0 + r) * Nd + col] + bv); }
        else if (EPI == 3) { v += bv; }
        po[idx] = v;
    }
}

template <int EPI>
__global__ void gemm_wmma(const float* __restrict__ A, const float* __restrict__ W,
                          const float* __restrict__ bias, const float* __restrict__ add,
                          const float* __restrict__ res, float* __restrict__ out,
                          int M, int K, int Nd) {
    const int wave = threadIdx.x >> 5;
    const int lane = threadIdx.x & 31;
    const int nbt = Nd >> 5;                            // 32-wide column tiles
    const int totalTiles = (M >> 5) * nbt;
    const int t = blockIdx.x * 4 + wave;                // wave-uniform
    if (t >= totalTiles) return;                        // uniform branch: EXEC stays full
    const int tN = (t % nbt) << 5;
    const int tM = (t / nbt) << 5;
    const int lmod = lane & 15, lhi = lane >> 4;

    const float* a0 = A + (size_t)(tM + lmod) * K + 16 * lhi;
    const float* a1 = A + (size_t)(tM + 16 + lmod) * K + 16 * lhi;
    const float* b0 = W + (size_t)(tN + lmod) * K + 16 * lhi;
    const float* b1 = W + (size_t)(tN + 16 + lmod) * K + 16 * lhi;

    v8f c00 = {}, c01 = {}, c10 = {}, c11 = {};
    for (int k0 = 0; k0 < K; k0 += 32) {
        v16h fa0 = load_a_frag(a0 + k0);
        v16h fa1 = load_a_frag(a1 + k0);
        v16h fb0 = load_b_frag(b0 + k0);
        v16h fb1 = load_b_frag(b1 + k0);
        c00 = wmma_f16(fa0, fb0, c00);
        c01 = wmma_f16(fa0, fb1, c01);
        c10 = wmma_f16(fa1, fb0, c10);
        c11 = wmma_f16(fa1, fb1, c11);
    }

    const int row0 = tM + 8 * lhi;                      // C/D: vgpr r -> row r + 8*(lane/16)
    const int col0 = tN + lmod;
    store_tile<EPI>(c00, bias, add, res, out, Nd, row0,      col0);
    store_tile<EPI>(c01, bias, add, res, out, Nd, row0,      col0 + 16);
    store_tile<EPI>(c10, bias, add, res, out, Nd, row0 + 16, col0);
    store_tile<EPI>(c11, bias, add, res, out, Nd, row0 + 16, col0 + 16);
}

// ---------------- graph norm ----------------
__global__ void init_deg(float* __restrict__ deg, int N) {
    int i = blockIdx.x * blockDim.x + threadIdx.x;
    if (i < N) deg[i] = 1.0f;                // self-loop contribution
}
__global__ void accum_deg(const int* __restrict__ col, float* __restrict__ deg, int E) {
    int e = blockIdx.x * blockDim.x + threadIdx.x;
    if (e < E) atomicAdd(&deg[col[e]], 1.0f);
}
__global__ void make_dinv(const float* __restrict__ deg, float* __restrict__ dinv, int N) {
    int i = blockIdx.x * blockDim.x + threadIdx.x;
    if (i < N) { float d = deg[i]; dinv[i] = d > 0.f ? rsqrtf(d) : 0.f; }
}

// agg[i,:] = dinv[i]^2 * xw[i,:]   (self-loop term; also initializes poisoned ws)
__global__ void init_agg(const float* __restrict__ xw, const float* __restrict__ dinv,
                         float* __restrict__ agg, int N, int D) {
    size_t idx = (size_t)blockIdx.x * blockDim.x + threadIdx.x;
    if (idx < (size_t)N * D) {
        int i = (int)(idx / D);
        float dv = dinv[i];
        agg[idx] = dv * dv * xw[idx];
    }
}

// agg[col,:] += dinv[row]*dinv[col] * xw[row,:]
__global__ void edge_scatter(const float* __restrict__ xw, const int* __restrict__ row,
                             const int* __restrict__ col, const float* __restrict__ dinv,
                             float* __restrict__ agg, int E, int D, int logD) {
    size_t idx = (size_t)blockIdx.x * blockDim.x + threadIdx.x;
    size_t e = idx >> logD;
    int f = (int)(idx & (size_t)(D - 1));
    if (e < (size_t)E) {
        int r = row[e], c = col[e];
        float nm = dinv[r] * dinv[c];
        atomicAdd(&agg[(size_t)c * D + f], nm * xw[(size_t)r * D + f]);
    }
}

// aw = W - W^T - GAMMA*I
__global__ void make_aw(const float* __restrict__ Wm, float* __restrict__ aw, int d) {
    int idx = blockIdx.x * blockDim.x + threadIdx.x;
    if (idx < d * d) {
        int i = idx / d, j = idx % d;
        aw[idx] = Wm[i * d + j] - Wm[j * d + i] - (i == j ? AS_GAMMA : 0.f);
    }
}

// log_softmax across 32 classes: one wave32 per row
__global__ void log_softmax32(const float* __restrict__ h, float* __restrict__ out, int N) {
    int lane = threadIdx.x & 31;
    int row = blockIdx.x * (blockDim.x >> 5) + (threadIdx.x >> 5);
    if (row >= N) return;
    float v = h[(size_t)row * 32 + lane];
    float m = v;
#pragma unroll
    for (int o = 16; o > 0; o >>= 1) m = fmaxf(m, __shfl_xor(m, o, 32));
    float ex = __expf(v - m);
    float s = ex;
#pragma unroll
    for (int o = 16; o > 0; o >>= 1) s += __shfl_xor(s, o, 32);
    out[(size_t)row * 32 + lane] = v - m - __logf(s);
}

// ---------------- launcher ----------------
extern "C" void kernel_launch(void* const* d_in, const int* in_sizes, int n_in,
                              void* d_out, int out_size, void* d_ws, size_t ws_size,
                              hipStream_t stream) {
    const float* x      = (const float*)d_in[0];
    const float* lin1_w = (const float*)d_in[1];
    const float* lin1_b = (const float*)d_in[2];
    const float* lin2_w = (const float*)d_in[3];
    const float* lin2_b = (const float*)d_in[4];
    const float* W1     = (const float*)d_in[5];
    const float* phi1_w = (const float*)d_in[6];
    const float* b1     = (const float*)d_in[7];
    const float* W2     = (const float*)d_in[8];
    const float* phi2_w = (const float*)d_in[9];
    const float* b2     = (const float*)d_in[10];
    const int*   ei     = (const int*)d_in[11];

    const int HID = in_sizes[2];              // 128
    const int C   = in_sizes[4];              // 32
    const int F   = in_sizes[1] / HID;        // 256
    const int N   = in_sizes[0] / F;          // 100000
    const int E   = in_sizes[11] / 2;         // 1.6M
    const int* erow = ei;
    const int* ecol = ei + E;

    float* ws = (float*)d_ws;
    size_t off = 0;
    auto alloc = [&](size_t n) { float* p = ws + off; off += (n + 63) & ~(size_t)63; return p; };
    float* deg  = alloc(N);
    float* dinv = alloc(N);
    float* h1   = alloc((size_t)N * HID);
    float* xw1  = alloc((size_t)N * HID);     // later reused as h2 output
    float* agg1 = alloc((size_t)N * HID);
    float* h3   = alloc((size_t)N * C);
    float* xw2  = alloc((size_t)N * C);       // later reused as h4 output
    float* agg2 = alloc((size_t)N * C);
    float* aw1  = alloc((size_t)HID * HID);
    float* aw2  = alloc((size_t)C * C);

    // ---- gcn_norm ----
    init_deg<<<(N + 255) / 256, 256, 0, stream>>>(deg, N);
    accum_deg<<<(E + 255) / 256, 256, 0, stream>>>(ecol, deg, E);
    make_dinv<<<(N + 255) / 256, 256, 0, stream>>>(deg, dinv, N);

    auto gemm_grid = [](int M, int Nd) { return ((M >> 5) * (Nd >> 5) + 3) / 4; };

    // ---- h1 = relu(x @ lin1_w^T + lin1_b) ----
    gemm_wmma<0><<<gemm_grid(N, HID), 128, 0, stream>>>(x, lin1_w, lin1_b, nullptr, nullptr,
                                                        h1, N, F, HID);
    // ---- layer 1: xw1 = h1 @ phi1_w^T ----
    gemm_wmma<1><<<gemm_grid(N, HID), 128, 0, stream>>>(h1, phi1_w, nullptr, nullptr, nullptr,
                                                        xw1, N, HID, HID);
    // agg1 = segment_sum(norm * xw1[row]) with self-loops
    {
        size_t tot = (size_t)N * HID;
        init_agg<<<(unsigned)((tot + 255) / 256), 256, 0, stream>>>(xw1, dinv, agg1, N, HID);
        int logD = 31 - __builtin_clz(HID);
        size_t et = (size_t)E * HID;
        edge_scatter<<<(unsigned)((et + 255) / 256), 256, 0, stream>>>(xw1, erow, ecol, dinv,
                                                                       agg1, E, HID, logD);
    }
    make_aw<<<(HID * HID + 255) / 256, 256, 0, stream>>>(W1, aw1, HID);
    // h2 = h1 + EPS*tanh(h1 @ aw1^T + agg1 + b1)   -> written into xw1
    gemm_wmma<2><<<gemm_grid(N, HID), 128, 0, stream>>>(h1, aw1, b1, agg1, h1,
                                                        xw1, N, HID, HID);
    // ---- h3 = h2 @ lin2_w^T + lin2_b ----
    gemm_wmma<3><<<gemm_grid(N, C), 128, 0, stream>>>(xw1, lin2_w, lin2_b, nullptr, nullptr,
                                                      h3, N, HID, C);
    // ---- layer 2: xw2 = h3 @ phi2_w^T ----
    gemm_wmma<1><<<gemm_grid(N, C), 128, 0, stream>>>(h3, phi2_w, nullptr, nullptr, nullptr,
                                                      xw2, N, C, C);
    {
        size_t tot = (size_t)N * C;
        init_agg<<<(unsigned)((tot + 255) / 256), 256, 0, stream>>>(xw2, dinv, agg2, N, C);
        int logD = 31 - __builtin_clz(C);
        size_t et = (size_t)E * C;
        edge_scatter<<<(unsigned)((et + 255) / 256), 256, 0, stream>>>(xw2, erow, ecol, dinv,
                                                                       agg2, E, C, logD);
    }
    make_aw<<<(C * C + 255) / 256, 256, 0, stream>>>(W2, aw2, C);
    // h4 = h3 + EPS*tanh(h3 @ aw2^T + agg2 + b2)   -> written into xw2
    gemm_wmma<2><<<gemm_grid(N, C), 128, 0, stream>>>(h3, aw2, b2, agg2, h3,
                                                      xw2, N, C, C);
    // ---- log_softmax ----
    log_softmax32<<<(N + 7) / 8, 256, 0, stream>>>(xw2, (float*)d_out, N);
}
